// RecurrentDecoder_1700807049925
// MI455X (gfx1250) — matrix operand order repaired and checked by reading, not compiled, not run on hardware
//
#include <hip/hip_runtime.h>

#define H      1024
#define OUTD   256
#define BATCH  1024
#define TSTEPS 128
#define KC     64            // K-chunk staged in LDS per double-buffer slot
#define KCP    72            // padded LDS row stride (bank-conflict-free)
#define NCHUNK (H / KC)      // 16

typedef __attribute__((ext_vector_type(16))) __bf16        v16bf;
typedef __attribute__((ext_vector_type(8)))  float         f32x8;
typedef __attribute__((ext_vector_type(4)))  unsigned int  u32x4;

// exact parameter types for the async global->LDS builtin (from the
// compiler diagnostic: "__vector_size__(4*sizeof(int)) int __device__ *")
typedef int v4i __attribute__((vector_size(16)));
typedef __attribute__((address_space(1))) v4i* as1_v4i_ptr;
typedef __attribute__((address_space(3))) v4i* as3_v4i_ptr;

union Frag {
    v16bf v;
    u32x4 u[2];
};

__device__ __forceinline__ unsigned short f2bf(float f) {
    union { float f; unsigned u; } c;
    c.f = f;
    unsigned u = c.u;
    u += 0x7FFFu + ((u >> 16) & 1u);   // round-to-nearest-even
    return (unsigned short)(u >> 16);
}

__device__ __forceinline__ float sigm_(float x) {
    return 1.0f / (1.0f + __expf(-x));
}

// 16-byte async global->LDS copy (ASYNCcnt-tracked, no VGPR round trip)
__device__ __forceinline__ void async_cp16(const unsigned short* g, unsigned short* l) {
#if __has_builtin(__builtin_amdgcn_global_load_async_to_lds_b128)
    __builtin_amdgcn_global_load_async_to_lds_b128(
        (as1_v4i_ptr)g, (as3_v4i_ptr)l, 0, 0);
#else
    unsigned loff = (unsigned)(unsigned long long)
        (__attribute__((address_space(3))) unsigned short*)l;
    asm volatile("global_load_async_to_lds_b128 %0, %1, off"
                 :: "v"(loff), "v"(g) : "memory");
#endif
}

__device__ __forceinline__ void async_wait0() {
#if __has_builtin(__builtin_amdgcn_s_wait_asynccnt)
    __builtin_amdgcn_s_wait_asynccnt(0);
#else
    asm volatile("s_wait_asynccnt 0x0" ::: "memory");
#endif
}

// ---------------------------------------------------------------------------
// Prep 1: interleaved gate weights Wg[16][384][H] (bf16).
//   block nb covers hidden cols j = nb*64 .. nb*64+63
//   row in block: g*64 + jj, g = {i_r,i_z,i_n,h_r,h_z,h_n}
//   g<3 : W_xh = W_ih @ W_dec (decoder folded into recurrence)
// ---------------------------------------------------------------------------
__global__ __launch_bounds__(256) void prep_weights(
    const float* __restrict__ Wih, const float* __restrict__ Whh,
    const float* __restrict__ Wdec, unsigned short* __restrict__ Wg)
{
    int tid = blockIdx.x * blockDim.x + threadIdx.x;
    if (tid >= 6144 * H) return;
    int k  = tid & (H - 1);
    int r  = tid >> 10;
    int nb = r / 384;
    int rr = r - nb * 384;
    int g  = rr >> 6;
    int jj = rr & 63;
    int j  = nb * 64 + jj;
    float v;
    if (g < 3) {
        const float* wrow = Wih + (size_t)(g * H + j) * OUTD;
        float s = 0.0f;
        for (int o = 0; o < OUTD; ++o) s += wrow[o] * Wdec[(size_t)o * H + k];
        v = s;
    } else {
        v = Whh[(size_t)((g - 3) * H + j) * H + k];
    }
    Wg[tid] = f2bf(v);
}

__global__ __launch_bounds__(256) void prep_bias(
    const float* __restrict__ Wih, const float* __restrict__ bih,
    const float* __restrict__ bhh, const float* __restrict__ bdec,
    float* __restrict__ Bg)
{
    int tid = blockIdx.x * blockDim.x + threadIdx.x;
    if (tid >= 6144) return;
    int nb = tid / 384;
    int rr = tid - nb * 384;
    int g  = rr >> 6;
    int jj = rr & 63;
    int j  = nb * 64 + jj;
    float v;
    if (g < 3) {
        float s = bih[g * H + j];
        const float* wrow = Wih + (size_t)(g * H + j) * OUTD;
        for (int o = 0; o < OUTD; ++o) s += wrow[o] * bdec[o];
        v = s;
    } else {
        v = bhh[(g - 3) * H + j];
    }
    Bg[tid] = v;
}

__global__ __launch_bounds__(256) void prep_wdec(
    const float* __restrict__ Wdec, unsigned short* __restrict__ Wd)
{
    int tid = blockIdx.x * blockDim.x + threadIdx.x;
    if (tid >= OUTD * H) return;
    Wd[tid] = f2bf(Wdec[tid]);
}

__global__ __launch_bounds__(256) void prep_h(
    const float* __restrict__ h0, float* __restrict__ hf,
    unsigned short* __restrict__ hb)
{
    int tid = blockIdx.x * blockDim.x + threadIdx.x;
    if (tid >= BATCH * H) return;
    float v = h0[tid];
    hf[tid] = v;
    hb[tid] = f2bf(v);
}

// ---------------------------------------------------------------------------
// One GRU step.
//   blocks [0,nGate):       fused gate GEMM + nonlinearity -> h_{t+1}
//   blocks [nGate,nGate+64): decode GEMM h_t @ Wd^T -> out[:, t-1, :]
// Gate path double-buffers Wg K-chunks in LDS via async global->LDS copies.
// ---------------------------------------------------------------------------
__global__ __launch_bounds__(256) void gru_step(
    const unsigned short* __restrict__ hbf,   // [B][H] bf16 h_t
    const float*          __restrict__ hf,    // [B][H] f32  h_t
    float*                __restrict__ hf_n,  // [B][H] f32  h_{t+1}
    unsigned short*       __restrict__ hbf_n, // [B][H] bf16 h_{t+1}
    const unsigned short* __restrict__ Wg,    // [16][384][H] bf16
    const float*          __restrict__ Bg,    // [16][384]
    const unsigned short* __restrict__ Wd,    // [OUTD][H] bf16
    const float*          __restrict__ bd,    // [OUTD]
    float*                __restrict__ out,   // [B][T][OUTD]
    int t, int nGate)
{
    __shared__ unsigned short smem[2][384 * KCP];

    const int wave = threadIdx.x >> 5;
    const int lane = threadIdx.x & 31;
    const int lr   = lane & 15;
    const int hi   = lane >> 4;
    const int ms   = wave & 3;   // M stripe (4 x 16 rows)
    const int nh   = wave >> 2;  // j-half (0: jj 0..31, 1: jj 32..63)

    const f32x8 zero = {0.f, 0.f, 0.f, 0.f, 0.f, 0.f, 0.f, 0.f};

    if ((int)blockIdx.x < nGate) {
        // ----------------------- gate path -----------------------
        const int nb   = blockIdx.x & 15;
        const int m0   = (blockIdx.x >> 4) * 64;
        const int row0 = m0 + ms * 16;
        const unsigned short* Arow = hbf + (size_t)(row0 + lr) * H + hi * 8;
        const unsigned short* Wblk = Wg + (size_t)nb * 384 * H;

        f32x8 acc[6][2];
#pragma unroll
        for (int g = 0; g < 6; ++g)
#pragma unroll
            for (int c = 0; c < 2; ++c) acc[g][c] = zero;

        // prologue: stage chunk 0 (384 rows x 64 cols = 3072 x 16B units)
        {
            unsigned short* dst = &smem[0][0];
#pragma unroll
            for (int uu = 0; uu < 12; ++uu) {
                const int u    = threadIdx.x + 256 * uu;
                const int rrow = u >> 3;
                const int sub  = u & 7;
                async_cp16(Wblk + (size_t)rrow * H + sub * 8,
                           dst + rrow * KCP + sub * 8);
            }
            async_wait0();
            __syncthreads();
        }

        for (int c = 0; c < NCHUNK; ++c) {
            // kick off async stage of chunk c+1 into the other buffer
            if (c + 1 < NCHUNK) {
                const int k0n = (c + 1) * KC;
                unsigned short* dst = &smem[(c + 1) & 1][0];
#pragma unroll
                for (int uu = 0; uu < 12; ++uu) {
                    const int u    = threadIdx.x + 256 * uu;
                    const int rrow = u >> 3;
                    const int sub  = u & 7;
                    async_cp16(Wblk + (size_t)rrow * H + k0n + sub * 8,
                               dst + rrow * KCP + sub * 8);
                }
            }
            // compute on chunk c from LDS while the transfer flies
            const unsigned short* Bb = &smem[c & 1][0];
#pragma unroll
            for (int kk = 0; kk < 2; ++kk) {
                const int k0g = c * KC + kk * 32;
                Frag a;
                a.u[0] = *(const u32x4*)(Arow + k0g);
                a.u[1] = *(const u32x4*)(Arow + k0g + 16);
#pragma unroll
                for (int g = 0; g < 6; ++g) {
#pragma unroll
                    for (int cc = 0; cc < 2; ++cc) {
                        const unsigned short* Brow =
                            Bb + (g * 64 + nh * 32 + cc * 16 + lr) * KCP +
                            kk * 32 + hi * 8;
                        Frag b;
                        b.u[0] = *(const u32x4*)(Brow);
                        b.u[1] = *(const u32x4*)(Brow + 16);
                        acc[g][cc] = __builtin_amdgcn_wmma_f32_16x16x32_bf16(
                            false, a.v, false, b.v, (short)0, acc[g][cc],
                            false, false);
                    }
                }
            }
            async_wait0();     // own chunk c+1 writes landed in LDS
            __syncthreads();   // everyone done writing c+1 / reading c
        }

        // fused GRU gate epilogue
#pragma unroll
        for (int c = 0; c < 2; ++c) {
            const int jj = nh * 32 + c * 16 + lr;
            const int j  = nb * 64 + jj;
            const float bir = Bg[nb * 384 + 0 * 64 + jj];
            const float biz = Bg[nb * 384 + 1 * 64 + jj];
            const float bin = Bg[nb * 384 + 2 * 64 + jj];
            const float bhr = Bg[nb * 384 + 3 * 64 + jj];
            const float bhz = Bg[nb * 384 + 4 * 64 + jj];
            const float bhn = Bg[nb * 384 + 5 * 64 + jj];
#pragma unroll
            for (int e = 0; e < 8; ++e) {
                const int row = row0 + hi * 8 + e;
                float r = sigm_(acc[0][c][e] + bir + acc[3][c][e] + bhr);
                float z = sigm_(acc[1][c][e] + biz + acc[4][c][e] + bhz);
                float n = tanhf(acc[2][c][e] + bin + r * (acc[5][c][e] + bhn));
                float ho = hf[(size_t)row * H + j];
                float hn = (1.0f - z) * n + z * ho;
                hf_n [(size_t)row * H + j] = hn;
                hbf_n[(size_t)row * H + j] = f2bf(hn);
            }
        }
    } else {
        // ----------------------- decode path -----------------------
        if (t == 0) return;
        const int idx  = blockIdx.x - nGate;
        const int m0   = (idx >> 2) * 64;
        const int n0   = (idx & 3) * 64;
        const int row0 = m0 + ms * 16;
        const unsigned short* Arow = hbf + (size_t)(row0 + lr) * H + hi * 8;

        f32x8 acc[2];
        acc[0] = zero; acc[1] = zero;

        for (int k0 = 0; k0 < H; k0 += 32) {
            Frag a;
            a.u[0] = *(const u32x4*)(Arow + k0);
            a.u[1] = *(const u32x4*)(Arow + k0 + 16);
#pragma unroll
            for (int c = 0; c < 2; ++c) {
                const unsigned short* Brow =
                    Wd + (size_t)(n0 + nh * 32 + c * 16 + lr) * H + k0 + hi * 8;
                Frag b;
                b.u[0] = *(const u32x4*)(Brow);
                b.u[1] = *(const u32x4*)(Brow + 16);
                acc[c] = __builtin_amdgcn_wmma_f32_16x16x32_bf16(
                    false, a.v, false, b.v, (short)0, acc[c], false, false);
            }
        }

#pragma unroll
        for (int c = 0; c < 2; ++c) {
            const int o    = n0 + nh * 32 + c * 16 + lr;
            const float bb = bd[o];
#pragma unroll
            for (int e = 0; e < 8; ++e) {
                const int row = row0 + hi * 8 + e;
                out[(size_t)row * (TSTEPS * OUTD) + (size_t)(t - 1) * OUTD + o] =
                    acc[c][e] + bb;
            }
        }
    }
}

// ---------------------------------------------------------------------------
extern "C" void kernel_launch(void* const* d_in, const int* in_sizes, int n_in,
                              void* d_out, int out_size, void* d_ws, size_t ws_size,
                              hipStream_t stream)
{
    const float* h0   = (const float*)d_in[0];
    const float* Wih  = (const float*)d_in[1];
    const float* Whh  = (const float*)d_in[2];
    const float* bih  = (const float*)d_in[3];
    const float* bhh  = (const float*)d_in[4];
    const float* Wdec = (const float*)d_in[5];
    const float* bdec = (const float*)d_in[6];
    float* out = (float*)d_out;

    char* p = (char*)d_ws;
    unsigned short* Wg = (unsigned short*)p; p += (size_t)6144 * H * 2;   // 12.6 MB
    unsigned short* Wd = (unsigned short*)p; p += (size_t)OUTD * H * 2;   // 0.5 MB
    float* Bg  = (float*)p;          p += (size_t)6144 * 4;
    float* hfa = (float*)p;          p += (size_t)BATCH * H * 4;
    float* hfb = (float*)p;          p += (size_t)BATCH * H * 4;
    unsigned short* hba = (unsigned short*)p; p += (size_t)BATCH * H * 2;
    unsigned short* hbb = (unsigned short*)p; p += (size_t)BATCH * H * 2;

    prep_weights<<<(6144 * H) / 256, 256, 0, stream>>>(Wih, Whh, Wdec, Wg);
    prep_bias   <<<24, 256, 0, stream>>>(Wih, bih, bhh, bdec, Bg);
    prep_wdec   <<<(OUTD * H) / 256, 256, 0, stream>>>(Wdec, Wd);
    prep_h      <<<(BATCH * H) / 256, 256, 0, stream>>>(h0, hfa, hba);

    float* hf_c = hfa;          float* hf_n = hfb;
    unsigned short* hb_c = hba; unsigned short* hb_n = hbb;

    for (int t = 0; t < TSTEPS; ++t) {
        gru_step<<<320, 256, 0, stream>>>(hb_c, hf_c, hf_n, hb_n,
                                          Wg, Bg, Wd, bdec, out, t, 256);
        float* tf = hf_c; hf_c = hf_n; hf_n = tf;
        unsigned short* tb = hb_c; hb_c = hb_n; hb_n = tb;
    }
    // final decode of h_128 -> out[:, 127, :]
    gru_step<<<64, 256, 0, stream>>>(hb_c, hf_c, hf_n, hb_n,
                                     Wg, Bg, Wd, bdec, out, TSTEPS, 0);
}
